// DeltaNet_61246233641421
// MI455X (gfx1250) — matrix operand order repaired and checked by reading, not compiled, not run on hardware
//
#include <hip/hip_runtime.h>

// ---------------- problem constants ----------------
#define HID   1024
#define NH    8
#define NK    4
#define DD    128
#define BB    2
#define TT    2048
#define BT    (BB*TT)          // 4096
#define CHK   16               // scan chunk length
#define NCHK  (TT/CHK)         // 128
#define CLAMP_MAX 0.999995f

// ---------------- vector types (POD ext vectors only) ----------------
typedef __attribute__((ext_vector_type(16))) __bf16        v16bf;
typedef __attribute__((ext_vector_type(8)))  float         v8f;
typedef __attribute__((ext_vector_type(4)))  unsigned int  u4;
typedef __attribute__((ext_vector_type(2)))  unsigned int  u2;
typedef __attribute__((ext_vector_type(4)))  float         f4;
typedef __attribute__((ext_vector_type(4)))  unsigned int  u32x4;
typedef __attribute__((ext_vector_type(8)))  int           i32x8;
typedef __attribute__((ext_vector_type(4)))  int           i32x4;

union BFr { v16bf v; u4 u[2]; };

__device__ inline unsigned short f2bf(float f) {
    unsigned u = __float_as_uint(f);
    unsigned r = u + 0x7FFFu + ((u >> 16) & 1u);
    return (unsigned short)(r >> 16);
}
__device__ inline float bf2f(unsigned short s) {
    return __uint_as_float(((unsigned)s) << 16);
}
__device__ inline unsigned int pk2(float lo, float hi) {
    return (unsigned)f2bf(lo) | ((unsigned)f2bf(hi) << 16);
}

__device__ inline v8f wmma_bf(u4 a0, u4 a1, u4 b0, u4 b1, v8f c) {
    BFr a, b; a.u[0] = a0; a.u[1] = a1; b.u[0] = b0; b.u[1] = b1;
    return __builtin_amdgcn_wmma_f32_16x16x32_bf16(false, a.v, false, b.v,
                                                   (short)0, c, false, false);
}

// pack 16 contiguous f32 from LDS into a bf16 B-fragment half-pair
__device__ inline void pack16f(const float* p, u4& b0, u4& b1) {
    const f4* q = (const f4*)p;
    f4 f0 = q[0], f1 = q[1], f2 = q[2], f3 = q[3];
    b0 = (u4){pk2(f0.x, f0.y), pk2(f0.z, f0.w), pk2(f1.x, f1.y), pk2(f1.z, f1.w)};
    b1 = (u4){pk2(f2.x, f2.y), pk2(f2.z, f2.w), pk2(f3.x, f3.y), pk2(f3.z, f3.w)};
}

// ---------------- TDM: 1-D contiguous bf16 tile -> LDS, padded rows ----------------
// pad_enable inserts 4 dwords (8 halves) after every 64 dwords (128 halves):
// LDS row stride becomes 136 halves, matching our staged tile layout.
// 6-arg builtin form (clang-23 / therock headers): (g0, g1, g2, g3, g4, cpol).
__device__ inline void tdm_load_tile(unsigned lds_off, const unsigned short* gp,
                                     int nelem, int do_pad) {
    unsigned long long ga = (unsigned long long)(const void*)gp;
    u32x4 g0;
    g0[0] = 1u;                                  // count=1, user descriptor
    g0[1] = lds_off;                             // LDS byte address
    g0[2] = (unsigned)ga;                        // global_addr[31:0]
    g0[3] = ((unsigned)(ga >> 32) & 0x1FFFFFFu) | (2u << 30); // addr[56:32] | type=2
    i32x8 g1;
    unsigned w0 = (1u << 16);                    // data_size = 2 bytes
    if (do_pad) w0 |= (1u << 20) | (5u << 22) | (3u << 25); // pad: every 64 dw, +4 dw
    g1[0] = (int)w0;
    g1[1] = (int)(((unsigned)nelem & 0xFFFFu) << 16);        // tensor_dim0 lo16
    g1[2] = (int)((((unsigned)nelem >> 16) & 0xFFFFu) | (1u << 16)); // dim0 hi | dim1=1
    g1[3] = (int)(((unsigned)nelem & 0xFFFFu) << 16);        // tile_dim0
    g1[4] = 1;                                   // tile_dim1 = 1
    g1[5] = nelem;                               // tensor_dim0_stride lo32
    g1[6] = 0; g1[7] = 0;
    i32x4 g2 = {0, 0, 0, 0};
    i32x4 g3 = {0, 0, 0, 0};
    i32x8 g4 = {0, 0, 0, 0, 0, 0, 0, 0};
    __builtin_amdgcn_tensor_load_to_lds(g0, g1, g2, g3, g4, 0);
}

// ---------------- f32 -> bf16 conversion (x, row-major kept) ----------------
__global__ __launch_bounds__(256) void k_cvt(const float* __restrict__ src,
                                             unsigned short* __restrict__ dst, int n) {
    int i = (blockIdx.x * 256 + threadIdx.x) * 4;
    if (i < n) {
        f4 v = *(const f4*)&src[i];
        u2 p = {pk2(v.x, v.y), pk2(v.z, v.w)};
        *(u2*)&dst[i] = p;
    }
}

// ---------------- weight conversion + transpose: W[k][n] f32 -> WT[n][k] bf16 ------
// One block per output row n; thread t handles k = 4t..4t+3.
__global__ __launch_bounds__(256) void k_cvtw(const float* __restrict__ W,
                                              unsigned short* __restrict__ wt) {
    const int n = blockIdx.x, k4 = threadIdx.x * 4;
    float a = W[(size_t)(k4 + 0) * HID + n];
    float b = W[(size_t)(k4 + 1) * HID + n];
    float c = W[(size_t)(k4 + 2) * HID + n];
    float d = W[(size_t)(k4 + 3) * HID + n];
    u2 p = {pk2(a, b), pk2(c, d)};
    *(u2*)&wt[(size_t)n * HID + k4] = p;
}

// ---------------- Q/K/V projections: bf16 WMMA GEMM + elu1 ----------------
// grid (256 mtiles, 8 head-blocks, 3 weights), block 256 = 8 waves; wave w owns 16 cols.
// Weights are pre-transposed: wt[n][k], so B-tiles stage as straight b128 copies.
__global__ __launch_bounds__(256) void k_proj(const unsigned short* __restrict__ xbf,
        const unsigned short* __restrict__ wqt, const unsigned short* __restrict__ wkt,
        const unsigned short* __restrict__ wvt,
        unsigned short* __restrict__ qb, unsigned short* __restrict__ kb,
        unsigned short* __restrict__ vb) {
    __shared__ __align__(16) unsigned short xs[16 * 40];
    __shared__ __align__(16) unsigned short wT[128 * 40];
    const int tid = threadIdx.x, lane = tid & 31, wvi = tid >> 5;
    const int n = lane & 15, hl = lane >> 4;
    const int row0 = blockIdx.x * 16, n0 = blockIdx.y * 128, which = blockIdx.z;
    const unsigned short* WT = (which == 0) ? wqt : (which == 1) ? wkt : wvt;
    v8f acc = {};
    const int xm = tid >> 2, xq = tid & 3;       // x-tile: 64 active threads
    const int wc = tid >> 1, wh = tid & 1;       // W-tile: 128 rows x 2 chunks
    for (int kk = 0; kk < HID; kk += 32) {
        if (tid < 64)                            // stage x tile 16x32 (stride 40)
            *(u4*)&xs[xm * 40 + xq * 8] =
                *(const u4*)&xbf[(size_t)(row0 + xm) * HID + kk + xq * 8];
        // stage W^T tile 128x16 (stride 40): 1 b128 load + 1 b128 store per thread
        *(u4*)&wT[wc * 40 + wh * 8] =
            *(const u4*)&WT[(size_t)(n0 + wc) * HID + kk + wh * 8];
        __syncthreads();
        u4 a0 = *(const u4*)&xs[n * 40 + 8 * hl];
        u4 a1 = *(const u4*)&xs[n * 40 + 16 + 8 * hl];
        u4 b0 = *(const u4*)&wT[(wvi * 16 + n) * 40 + 8 * hl];
        u4 b1 = *(const u4*)&wT[(wvi * 16 + n) * 40 + 16 + 8 * hl];
        acc = wmma_bf(a0, a1, b0, b1, acc);
        __syncthreads();
    }
    const int d = wvi * 16 + n, hh = blockIdx.y;
    unsigned short* dst = (which == 0) ? qb : (which == 1) ? kb : vb;
#pragma unroll
    for (int i = 0; i < 8; ++i) {
        int m = i + 8 * hl, bt = row0 + m, b = bt >> 11, t = bt & (TT - 1);
        float z = acc[i];
        if (which < 2) z = (z > 0.f) ? z + 1.f : __expf(z);   // elu(z)+1
        dst[((size_t)((b * NH + hh) * TT + t)) * DD + d] = f2bf(z);
    }
}

// ---------------- beta / mix(softmax) projections ----------------
__global__ __launch_bounds__(256) void k_small(const float* __restrict__ x,
        const float* __restrict__ Wb, const float* __restrict__ bbv,
        const float* __restrict__ Wm, const float* __restrict__ bm,
        float* __restrict__ beta, float* __restrict__ wmix) {
    __shared__ __align__(16) float xr[HID];
    __shared__ float mixsh[NH * NK];
    const int bt = blockIdx.x, tid = threadIdx.x, lane = tid & 31, wvi = tid >> 5;
    *(f4*)&xr[tid * 4] = *(const f4*)&x[(size_t)bt * HID + tid * 4];
    __syncthreads();
    for (int jj = wvi; jj < NH + NH * NK; jj += 8) {
        const float* Wcol; int stride; float bias;
        if (jj < NH) { Wcol = Wb + jj; stride = NH; bias = bbv[jj]; }
        else         { Wcol = Wm + (jj - NH); stride = NH * NK; bias = bm[jj - NH]; }
        float s = 0.f;
        for (int i = 0; i < HID / 32; ++i) {
            int di = lane + 32 * i;
            s += xr[di] * Wcol[(size_t)di * stride];
        }
        for (int off = 16; off > 0; off >>= 1) s += __shfl_xor(s, off);
        if (lane == 0) {
            float z = s + bias;
            if (jj < NH) beta[(size_t)bt * NH + jj] = 1.f / (1.f + __expf(-z));
            else         mixsh[jj - NH] = z;
        }
    }
    __syncthreads();
    if (tid < NH) {
        float m0 = mixsh[tid*4], m1 = mixsh[tid*4+1], m2 = mixsh[tid*4+2], m3 = mixsh[tid*4+3];
        float mx = fmaxf(fmaxf(m0, m1), fmaxf(m2, m3));
        float e0 = __expf(m0-mx), e1 = __expf(m1-mx), e2 = __expf(m2-mx), e3 = __expf(m3-mx);
        float inv = 1.f / (e0 + e1 + e2 + e3);
        wmix[(size_t)bt*32 + tid*4 + 0] = e0*inv; wmix[(size_t)bt*32 + tid*4 + 1] = e1*inv;
        wmix[(size_t)bt*32 + tid*4 + 2] = e2*inv; wmix[(size_t)bt*32 + tid*4 + 3] = e3*inv;
    }
}

// ---------------- chunked decay scan: 64 WGs = (b,h,k), 8 waves own 16-col strips ----
__global__ __launch_bounds__(256) void k_scan(const unsigned short* __restrict__ qbuf,
        const unsigned short* __restrict__ kbuf, const unsigned short* __restrict__ vbuf,
        const float* __restrict__ beta, const float* __restrict__ mask,
        const float* __restrict__ base_logit, unsigned short* __restrict__ osc) {
    __shared__ __align__(16) float          S_sh[DD * 132];      // state, [e][d], pad 4
    __shared__ __align__(16) unsigned short qc_sh[CHK * 136];    // raw q chunk  (TDM pad)
    __shared__ __align__(16) unsigned short kc_sh[CHK * 136];    // raw k chunk  (TDM pad)
    __shared__ __align__(16) unsigned short vc_sh[CHK * 136];    // raw v chunk  (TDM pad)
    __shared__ __align__(16) unsigned short qdec_sh[CHK * 136];  // q * cum decay
    __shared__ __align__(16) unsigned short kdT_sh[DD * 24];     // (k*suf)^T [d][t]
    __shared__ __align__(16) unsigned short vT_sh[DD * 24];      // v^T [e][t]
    __shared__ __align__(16) unsigned short aw_sh[8 * 16 * 24];  // per-wave masked QK^T
    __shared__ float rr_sh[CHK], cum_sh[CHK], suf_sh[CHK], mv_sh[CHK];

    const int tid = threadIdx.x, lane = tid & 31, wv = tid >> 5;
    const int n = lane & 15, hl = lane >> 4, e0 = wv * 16;
    const int wg = blockIdx.x, kk = wg & 3, h = (wg >> 2) & 7, b = wg >> 5;
    const float brk = 1.f / (1.f + __expf(-base_logit[h * NK + kk]));
    const size_t chbase = (size_t)((b * NH + h) * TT) * DD;      // q/k/v base
    const size_t obase  = (size_t)(((b * NH + h) * NK + kk) * TT) * DD;
    const u4 z4 = {0, 0, 0, 0};

    for (int i = tid; i < DD * 132; i += 256) S_sh[i] = 0.f;     // S = 0
    __syncthreads();

    for (int ci = 0; ci < NCHK; ++ci) {
        const int t0 = ci * CHK;
        if (tid == 0) {  // TDM-stage the three 16x128 bf16 tiles (padded to stride 136)
            tdm_load_tile((unsigned)(size_t)&qc_sh[0], qbuf + chbase + (size_t)t0 * DD, CHK * DD, 1);
            tdm_load_tile((unsigned)(size_t)&kc_sh[0], kbuf + chbase + (size_t)t0 * DD, CHK * DD, 1);
            tdm_load_tile((unsigned)(size_t)&vc_sh[0], vbuf + chbase + (size_t)t0 * DD, CHK * DD, 1);
            __builtin_amdgcn_s_wait_tensorcnt(0);
        }
        if (tid < CHK) {  // per-step effective decay
            float mval = mask[(size_t)b * TT + t0 + tid];
            float be = beta[(size_t)(b * TT + t0 + tid) * NH + h];
            float r = fminf(fmaxf(be * brk, 0.f), CLAMP_MAX);
            rr_sh[tid] = (mval > 0.f) ? r : 1.0f;
            mv_sh[tid] = (mval > 0.f) ? 1.f : 0.f;
        }
        __syncthreads();
        if (tid < CHK) {  // prefix / suffix decay products
            float c = 1.f, s = 1.f;
            for (int p = 0; p <= tid; ++p) c *= rr_sh[p];
            for (int p = tid + 1; p < CHK; ++p) s *= rr_sh[p];
            cum_sh[tid] = c; suf_sh[tid] = s;
        }
        __syncthreads();
        for (int idx = tid; idx < CHK * DD; idx += 256) {  // build qdec / kdec^T / v^T
            int t = idx >> 7, d = idx & 127;
            float qv = bf2f(qc_sh[t * 136 + d]);
            float kv = bf2f(kc_sh[t * 136 + d]);
            float vv = bf2f(vc_sh[t * 136 + d]);
            qdec_sh[t * 136 + d] = f2bf(qv * cum_sh[t]);
            kdT_sh[d * 24 + t]   = f2bf(kv * suf_sh[t] * mv_sh[t]);
            vT_sh[d * 24 + t]    = f2bf(vv);
        }
        __syncthreads();

        const float total = cum_sh[CHK - 1];
        // ---- A = Q K^T (f32 accum), 4 WMMAs over d ----
        v8f aqk = {};
        for (int j = 0; j < 4; ++j) {
            u4 a0 = *(const u4*)&qc_sh[n * 136 + 32 * j + 8 * hl];
            u4 a1 = *(const u4*)&qc_sh[n * 136 + 32 * j + 16 + 8 * hl];
            u4 b0 = *(const u4*)&kc_sh[n * 136 + 32 * j + 8 * hl];
            u4 b1 = *(const u4*)&kc_sh[n * 136 + 32 * j + 16 + 8 * hl];
            aqk = wmma_bf(a0, a1, b0, b1, aqk);
        }
        // ---- causal decay mask, to bf16 tile aw[t][s] ----
#pragma unroll
        for (int i = 0; i < 8; ++i) {
            int t = i + 8 * hl;
            float wts = 0.f;
            if (n <= t) { wts = mv_sh[n]; for (int p = n + 1; p <= t; ++p) wts *= rr_sh[p]; }
            aw_sh[wv * 384 + t * 24 + n] = f2bf(aqk[i] * wts);
        }
        // ---- O = (A.W) V  +  Qdec S_old ----
        v8f ocr = {};
        u4 vb0 = *(const u4*)&vT_sh[(e0 + n) * 24 + 8 * hl];
        u4 vb1 = z4;
        {
            u4 a0 = *(const u4*)&aw_sh[wv * 384 + n * 24 + 8 * hl];
            ocr = wmma_bf(a0, z4, vb0, vb1, ocr);
        }
        for (int j = 0; j < 4; ++j) {
            u4 a0 = *(const u4*)&qdec_sh[n * 136 + 32 * j + 16 * hl];
            u4 a1 = *(const u4*)&qdec_sh[n * 136 + 32 * j + 16 * hl + 8];
            u4 b0, b1;
            pack16f(&S_sh[(e0 + n) * 132 + 32 * j + 16 * hl], b0, b1);
            ocr = wmma_bf(a0, a1, b0, b1, ocr);
        }
#pragma unroll
        for (int i = 0; i < 8; ++i) {  // write per-k head output
            int t = i + 8 * hl;
            osc[obase + (size_t)(t0 + t) * DD + e0 + n] = f2bf(ocr[i] * mv_sh[t]);
        }
        // ---- S_new = total*S_old + Kdec^T V (8 tiles, C-accumulate WMMA) ----
        for (int r = 0; r < 8; ++r) {
            float* sp = &S_sh[(e0 + n) * 132 + 16 * r + 8 * hl];
            f4 s0 = *(const f4*)sp, s1 = *(const f4*)(sp + 4);
            v8f cin;
            cin[0] = s0.x * total; cin[1] = s0.y * total; cin[2] = s0.z * total; cin[3] = s0.w * total;
            cin[4] = s1.x * total; cin[5] = s1.y * total; cin[6] = s1.z * total; cin[7] = s1.w * total;
            u4 a0 = *(const u4*)&kdT_sh[(16 * r + n) * 24 + 8 * hl];
            v8f dn = wmma_bf(a0, z4, vb0, vb1, cin);
            *(f4*)sp       = (f4){dn[0], dn[1], dn[2], dn[3]};
            *(f4*)(sp + 4) = (f4){dn[4], dn[5], dn[6], dn[7]};
        }
        __syncthreads();   // protect shared tiles before next chunk's TDM
    }
}

// ---------------- per-head K-scale mixing ----------------
__global__ __launch_bounds__(256) void k_mix(const unsigned short* __restrict__ osc,
        const float* __restrict__ wmix, unsigned short* __restrict__ hc) {
    const int bt = blockIdx.x, c = threadIdx.x * 4;
    const int b = bt >> 11, t = bt & (TT - 1);
#pragma unroll
    for (int r = 0; r < 4; ++r) {
        int col = c + r, hh = col >> 7, d = col & 127;
        float acc = 0.f;
#pragma unroll
        for (int k2 = 0; k2 < NK; ++k2) {
            float w = wmix[(size_t)bt * 32 + hh * 4 + k2];
            float o = bf2f(osc[((size_t)((b * NH + hh) * NK + k2) * TT + t) * DD + d]);
            acc += w * o;
        }
        hc[(size_t)bt * HID + col] = f2bf(acc);
    }
}

// ---------------- output projection + residual (f32 out) ----------------
__global__ __launch_bounds__(256) void k_gemmout(const unsigned short* __restrict__ hc,
        const unsigned short* __restrict__ wot, const float* __restrict__ x,
        const float* __restrict__ bo, float* __restrict__ y) {
    __shared__ __align__(16) unsigned short xs[16 * 40];
    __shared__ __align__(16) unsigned short wT[128 * 40];
    const int tid = threadIdx.x, lane = tid & 31, wvi = tid >> 5;
    const int n = lane & 15, hl = lane >> 4;
    const int row0 = blockIdx.x * 16, n0 = blockIdx.y * 128;
    v8f acc = {};
    const int xm = tid >> 2, xq = tid & 3;
    const int wc = tid >> 1, wh = tid & 1;
    for (int kk = 0; kk < HID; kk += 32) {
        if (tid < 64)
            *(u4*)&xs[xm * 40 + xq * 8] =
                *(const u4*)&hc[(size_t)(row0 + xm) * HID + kk + xq * 8];
        *(u4*)&wT[wc * 40 + wh * 8] =
            *(const u4*)&wot[(size_t)(n0 + wc) * HID + kk + wh * 8];
        __syncthreads();
        u4 a0 = *(const u4*)&xs[n * 40 + 8 * hl];
        u4 a1 = *(const u4*)&xs[n * 40 + 16 + 8 * hl];
        u4 b0 = *(const u4*)&wT[(wvi * 16 + n) * 40 + 8 * hl];
        u4 b1 = *(const u4*)&wT[(wvi * 16 + n) * 40 + 16 + 8 * hl];
        acc = wmma_bf(a0, a1, b0, b1, acc);
        __syncthreads();
    }
    const int col = n0 + wvi * 16 + n;
#pragma unroll
    for (int i = 0; i < 8; ++i) {
        int bt = row0 + i + 8 * hl;
        y[(size_t)bt * HID + col] = acc[i] + x[(size_t)bt * HID + col] + bo[col];
    }
}

// ---------------- LayerNorm ----------------
__global__ __launch_bounds__(256) void k_ln(const float* __restrict__ y,
        const float* __restrict__ g, const float* __restrict__ be, float* __restrict__ out) {
    __shared__ float red[256];
    const int row = blockIdx.x, tid = threadIdx.x;
    f4 v = *(const f4*)&y[(size_t)row * HID + tid * 4];
    red[tid] = v.x + v.y + v.z + v.w;
    __syncthreads();
    for (int o = 128; o > 0; o >>= 1) { if (tid < o) red[tid] += red[tid + o]; __syncthreads(); }
    float mu = red[0] * (1.f / HID);
    __syncthreads();
    float d0 = v.x - mu, d1 = v.y - mu, d2 = v.z - mu, d3 = v.w - mu;
    red[tid] = d0 * d0 + d1 * d1 + d2 * d2 + d3 * d3;
    __syncthreads();
    for (int o = 128; o > 0; o >>= 1) { if (tid < o) red[tid] += red[tid + o]; __syncthreads(); }
    float inv = rsqrtf(red[0] * (1.f / HID) + 1e-5f);
    int c = tid * 4;
    out[(size_t)row * HID + c + 0] = d0 * inv * g[c + 0] + be[c + 0];
    out[(size_t)row * HID + c + 1] = d1 * inv * g[c + 1] + be[c + 1];
    out[(size_t)row * HID + c + 2] = d2 * inv * g[c + 2] + be[c + 2];
    out[(size_t)row * HID + c + 3] = d3 * inv * g[c + 3] + be[c + 3];
}

// ---------------- workspace layout ----------------
static constexpr size_t OFF_XBF  = 0;                                  // 8  MiB
static constexpr size_t OFF_WQ   = OFF_XBF + (size_t)BT * HID * 2;
static constexpr size_t OFF_WK   = OFF_WQ  + (size_t)HID * HID * 2;
static constexpr size_t OFF_WV   = OFF_WK  + (size_t)HID * HID * 2;
static constexpr size_t OFF_WO   = OFF_WV  + (size_t)HID * HID * 2;
static constexpr size_t OFF_Q    = OFF_WO  + (size_t)HID * HID * 2;
static constexpr size_t OFF_K    = OFF_Q   + (size_t)BT * HID * 2;
static constexpr size_t OFF_V    = OFF_K   + (size_t)BT * HID * 2;
static constexpr size_t OFF_BETA = OFF_V   + (size_t)BT * HID * 2;
static constexpr size_t OFF_WMIX = OFF_BETA + (size_t)BT * NH * 4;
static constexpr size_t OFF_OSC  = OFF_WMIX + (size_t)BT * NH * NK * 4;
static constexpr size_t OFF_HC   = OFF_OSC  + (size_t)BB * NH * NK * TT * DD * 2;
static constexpr size_t OFF_Y    = OFF_HC   + (size_t)BT * HID * 2;

extern "C" void kernel_launch(void* const* d_in, const int* in_sizes, int n_in,
                              void* d_out, int out_size, void* d_ws, size_t ws_size,
                              hipStream_t stream) {
    const float* x    = (const float*)d_in[0];
    const float* mask = (const float*)d_in[1];
    const float* Wq   = (const float*)d_in[2];
    const float* Wk   = (const float*)d_in[3];
    const float* Wv   = (const float*)d_in[4];
    const float* Wb   = (const float*)d_in[5];
    const float* bbv  = (const float*)d_in[6];
    const float* Wm   = (const float*)d_in[7];
    const float* bm   = (const float*)d_in[8];
    const float* Wo   = (const float*)d_in[9];
    const float* bo   = (const float*)d_in[10];
    const float* bl   = (const float*)d_in[11];
    const float* ln_g = (const float*)d_in[12];
    const float* ln_b = (const float*)d_in[13];
    float* out = (float*)d_out;

    char* ws = (char*)d_ws;
    unsigned short* xbf  = (unsigned short*)(ws + OFF_XBF);
    unsigned short* wqt  = (unsigned short*)(ws + OFF_WQ);
    unsigned short* wkt  = (unsigned short*)(ws + OFF_WK);
    unsigned short* wvt  = (unsigned short*)(ws + OFF_WV);
    unsigned short* wot  = (unsigned short*)(ws + OFF_WO);
    unsigned short* qb   = (unsigned short*)(ws + OFF_Q);
    unsigned short* kb   = (unsigned short*)(ws + OFF_K);
    unsigned short* vb   = (unsigned short*)(ws + OFF_V);
    float*          beta = (float*)(ws + OFF_BETA);
    float*          wmix = (float*)(ws + OFF_WMIX);
    unsigned short* osc  = (unsigned short*)(ws + OFF_OSC);
    unsigned short* hc   = (unsigned short*)(ws + OFF_HC);
    float*          ybuf = (float*)(ws + OFF_Y);

    k_cvt<<<(BT * HID) / 1024, 256, 0, stream>>>(x, xbf, BT * HID);
    k_cvtw<<<HID, 256, 0, stream>>>(Wq, wqt);
    k_cvtw<<<HID, 256, 0, stream>>>(Wk, wkt);
    k_cvtw<<<HID, 256, 0, stream>>>(Wv, wvt);
    k_cvtw<<<HID, 256, 0, stream>>>(Wo, wot);

    k_proj<<<dim3(BT / 16, NH, 3), 256, 0, stream>>>(xbf, wqt, wkt, wvt, qb, kb, vb);
    k_small<<<BT, 256, 0, stream>>>(x, Wb, bbv, Wm, bm, beta, wmix);
    k_scan<<<BB * NH * NK, 256, 0, stream>>>(qb, kb, vb, beta, mask, bl, osc);
    k_mix<<<BT, 256, 0, stream>>>(osc, wmix, hc);
    k_gemmout<<<dim3(BT / 16, NH), 256, 0, stream>>>(hc, wot, x, bo, ybuf);
    k_ln<<<BT, 256, 0, stream>>>(ybuf, ln_g, ln_b, out);
    (void)in_sizes; (void)n_in; (void)out_size; (void)ws_size;
}
